// TransformerAttn_27307402068445
// MI455X (gfx1250) — matrix operand order repaired and checked by reading, not compiled, not run on hardware
//
#include <hip/hip_runtime.h>
#include <hip/hip_bf16.h>

// ---------------------------------------------------------------------------
// TransformerAttn for MI455X (gfx1250, wave32, WMMA)
//   S=2048, B=16, D=512, fp32 in/out, bf16 WMMA with f32 accumulation.
//   Phase 0: one-shot f32->bf16 conversion of seq (to batch-major Xbf) and W.
//   Phase 1: V/Q/K projections (K stored transposed) -- pure bf16 WMMA GEMM.
//   Phase 2: flash-attention (online softmax) per (batch, 32-row block),
//            V A-operand register-resident across the whole j-loop.
// ---------------------------------------------------------------------------

#define SEQ_S 2048
#define SEQ_B 16
#define SEQ_D 512

typedef __bf16 bf16_t;
typedef __attribute__((ext_vector_type(2)))  __bf16 v2bf;
typedef __attribute__((ext_vector_type(16))) __bf16 v16bf;
typedef __attribute__((ext_vector_type(8)))  float  v8f;

// packed f32x2 -> bf16x2 (single v_cvt_pk_bf16_f32 when available)
__device__ __forceinline__ v2bf pack2(float lo, float hi) {
#if __has_builtin(__builtin_amdgcn_cvt_pk_bf16_f32)
    return __builtin_amdgcn_cvt_pk_bf16_f32(lo, hi);
#else
    v2bf r; r[0] = (__bf16)lo; r[1] = (__bf16)hi; return r;
#endif
}
__device__ __forceinline__ bf16_t f2bf(float f) { return (bf16_t)f; }

// A-operand (M x K = 16 x 32, bf16). Lane m(0..15): row m, K 0..7 / 16..23;
// lane m+16: same row, K 8..15 / 24..31.  `row` = this lane's row (K=0).
__device__ __forceinline__ v16bf load_a_tile(const bf16_t* row, int lane) {
    const int ko = (lane & 16) ? 8 : 0;
    union { uint4 u[2]; v16bf v; } t;
    t.u[0] = *reinterpret_cast<const uint4*>(row + ko);
    t.u[1] = *reinterpret_cast<const uint4*>(row + 16 + ko);
    return t.v;
}

// B-operand (K x N = 32 x 16, bf16), loaded as rows of the N x K transpose.
// Lanes 0-15: K 0..7 / 16..23, lanes 16-31: K 8..15 / 24..31.
__device__ __forceinline__ v16bf load_b_tile(const bf16_t* row, int lane) {
    const int ko = (lane & 16) ? 8 : 0;
    union { uint4 u[2]; v16bf v; } t;
    t.u[0] = *reinterpret_cast<const uint4*>(row + ko);
    t.u[1] = *reinterpret_cast<const uint4*>(row + 16 + ko);
    return t.v;
}

__device__ __forceinline__ v8f wmma_bf16(v16bf a, v16bf b, v8f c) {
    return __builtin_amdgcn_wmma_f32_16x16x32_bf16(
        false, a, false, b, (short)0, c, false, false);
}

// ---------------------------------------------------------------------------
// Phase 0a: seq [S,B,D] f32 -> Xbf [B,S,D] bf16 (batch-major transpose).
// ---------------------------------------------------------------------------
__global__ __launch_bounds__(256) void cvt_seq_kernel(
    const float* __restrict__ seq, bf16_t* __restrict__ Xbf)
{
    const unsigned tid = blockIdx.x * 256u + threadIdx.x;
    const size_t idx = (size_t)tid * 8;                 // linear into seq
    const unsigned d  = (unsigned)(idx & (SEQ_D - 1));
    const unsigned sb = (unsigned)(idx >> 9);           // s*B + b
    const unsigned b  = sb & (SEQ_B - 1);
    const unsigned s  = sb >> 4;

    float4 f0 = *reinterpret_cast<const float4*>(seq + idx);
    float4 f1 = *reinterpret_cast<const float4*>(seq + idx + 4);
    union { v2bf v[4]; uint4 u; } t;
    t.v[0] = pack2(f0.x, f0.y);
    t.v[1] = pack2(f0.z, f0.w);
    t.v[2] = pack2(f1.x, f1.y);
    t.v[3] = pack2(f1.z, f1.w);
    *reinterpret_cast<uint4*>(Xbf + ((size_t)b * SEQ_S + s) * SEQ_D + d) = t.u;
}

// ---------------------------------------------------------------------------
// Phase 0b: Wv/Wq/Wk [512,512] f32 -> Wbf [3,512,512] bf16.
// ---------------------------------------------------------------------------
__global__ __launch_bounds__(256) void cvt_w_kernel(
    const float* __restrict__ Wv, const float* __restrict__ Wq,
    const float* __restrict__ Wk, bf16_t* __restrict__ Wbf)
{
    const unsigned tid = blockIdx.x * 256u + threadIdx.x;   // 98304 total
    const unsigned w = tid >> 15;                           // 32768 per matrix
    const size_t r8 = (size_t)(tid & 32767u) * 8;
    const float* src = ((w == 0) ? Wv : (w == 1) ? Wq : Wk) + r8;

    float4 f0 = *reinterpret_cast<const float4*>(src);
    float4 f1 = *reinterpret_cast<const float4*>(src + 4);
    union { v2bf v[4]; uint4 u; } t;
    t.v[0] = pack2(f0.x, f0.y);
    t.v[1] = pack2(f0.z, f0.w);
    t.v[2] = pack2(f1.x, f1.y);
    t.v[3] = pack2(f1.z, f1.w);
    *reinterpret_cast<uint4*>(Wbf + (size_t)w * SEQ_D * SEQ_D + r8) = t.u;
}

// ---------------------------------------------------------------------------
// Phase 1: projections.  out[s,d'] = sum_k X[s,k]*W[d',k] + bias[d']
//   grid = (D/64, S/64, B*3), block = 128 (4 waves).
// ---------------------------------------------------------------------------
__global__ __launch_bounds__(128) void proj_kernel(
    const bf16_t* __restrict__ Xbf, const bf16_t* __restrict__ Wbf,
    const float* __restrict__ bv, const float* __restrict__ bq,
    const float* __restrict__ bk,
    bf16_t* __restrict__ Vbf, bf16_t* __restrict__ Qbf,
    bf16_t* __restrict__ Ktbf)
{
    const int lane = threadIdx.x & 31;
    const int wid  = threadIdx.x >> 5;
    const int lm   = lane & 15;
    const int hi   = lane >> 4;

    const int dtile = blockIdx.x;          // 0..7
    const int stile = blockIdx.y;          // 0..31
    const int b     = blockIdx.z / 3;      // 0..15
    const int w     = blockIdx.z % 3;      // 0:V 1:Q 2:K

    const bf16_t* W  = Wbf + (size_t)w * SEQ_D * SEQ_D;
    const float* bia = (w == 0) ? bv : (w == 1) ? bq : bk;

    const int rowbase = stile * 64 + wid * 16;
    const int colbase = dtile * 64;

    const bf16_t* arow = Xbf + ((size_t)b * SEQ_S + rowbase + lm) * SEQ_D;

    v8f acc[4] = {};

    for (int ks = 0; ks < 16; ++ks) {       // K = 512 in steps of 32
        const int k0 = ks * 32;
        v16bf a = load_a_tile(arow + k0, lane);
#pragma unroll
        for (int cs = 0; cs < 4; ++cs) {
            const bf16_t* brow = W + (size_t)(colbase + cs * 16 + lm) * SEQ_D + k0;
            v16bf bm = load_b_tile(brow, lane);
            acc[cs] = wmma_bf16(a, bm, acc[cs]);
        }
    }

#pragma unroll
    for (int cs = 0; cs < 4; ++cs) {
        const int col = colbase + cs * 16 + lm;
        const float bb = bia[col];
#pragma unroll
        for (int r = 0; r < 8; ++r) {
            const int row = rowbase + r + hi * 8;     // C-layout row
            const float val = acc[cs][r] + bb;
            if (w == 0)
                Vbf[((size_t)b * SEQ_S + row) * SEQ_D + col] = f2bf(val);
            else if (w == 1)
                Qbf[((size_t)b * SEQ_S + row) * SEQ_D + col] = f2bf(val);
            else
                Ktbf[((size_t)b * SEQ_D + col) * SEQ_S + row] = f2bf(val);
        }
    }
}

// ---------------------------------------------------------------------------
// Phase 2: flash attention.
//   grid = (S/32, B), block = 256 (8 waves: wr=wid>>2, wc=wid&3).
//   Per WG: 32 V-rows.  j-loop over 64-column Q blocks (32 iterations).
//   V strip (16x512) for the wave's score rows is register-resident (va[16]).
// ---------------------------------------------------------------------------
__global__ __launch_bounds__(256) void attn_kernel(
    const bf16_t* __restrict__ Vbf, const bf16_t* __restrict__ Qbf,
    const bf16_t* __restrict__ Ktbf, float* __restrict__ out)
{
    __shared__ float  Sls[32][68];     // score tile (padded)
    __shared__ bf16_t Pls[32][72];     // softmax'd P tile, bf16 (padded)
    __shared__ float  Mrow[32], Lrow[32], Alpha[32];
    __shared__ float  Pmax[32][8], Psum[32][8];

    const int t    = threadIdx.x;
    const int lane = t & 31;
    const int wid  = t >> 5;
    const int wr   = wid >> 2;   // 0..1
    const int wc   = wid & 3;    // 0..3
    const int lm   = lane & 15;
    const int hi   = lane >> 4;

    const int b       = blockIdx.y;
    const int rowbase = blockIdx.x * 32;

    if (t < 32) { Mrow[t] = -1e30f; Lrow[t] = 0.0f; }

    v8f o[8] = {};

    const bf16_t* Vb  = Vbf  + (size_t)b * SEQ_S * SEQ_D;
    const bf16_t* Qb  = Qbf  + (size_t)b * SEQ_S * SEQ_D;
    const bf16_t* Ktb = Ktbf + (size_t)b * SEQ_D * SEQ_S;

    // Register-resident A operand: this wave's 16 x 512 V strip (128 VGPRs).
    const bf16_t* vrow = Vb + (size_t)(rowbase + wr * 16 + lm) * SEQ_D;
    v16bf va[16];
#pragma unroll
    for (int ks = 0; ks < 16; ++ks) va[ks] = load_a_tile(vrow + ks * 32, lane);

    __syncthreads();

    for (int jt = 0; jt < SEQ_S / 64; ++jt) {
        const int jb = jt * 64;

        // ---- scores: one 16x16 f32 tile per wave, K = 512 ----
        v8f s = {};
        const bf16_t* qrow = Qb + (size_t)(jb + wc * 16 + lm) * SEQ_D;
#pragma unroll
        for (int ks = 0; ks < 16; ++ks) {
            v16bf bm = load_b_tile(qrow + ks * 32, lane);
            s = wmma_bf16(va[ks], bm, s);
        }

        __syncthreads();   // previous iteration done with Sls/Pls

#pragma unroll
        for (int r = 0; r < 8; ++r)
            Sls[wr * 16 + r + hi * 8][wc * 16 + lm] = s[r] * 0.04419417382f; // 1/sqrt(512)
        __syncthreads();

        // ---- online softmax over the 64 new columns ----
        // thread (wid=g, lane=r): row r, cols g*8 .. g*8+7
        float mloc = -1e30f;
#pragma unroll
        for (int c = 0; c < 8; ++c) mloc = fmaxf(mloc, Sls[lane][wid * 8 + c]);
        Pmax[lane][wid] = mloc;
        __syncthreads();

        if (t < 32) {
            float mt = Pmax[t][0];
#pragma unroll
            for (int g = 1; g < 8; ++g) mt = fmaxf(mt, Pmax[t][g]);
            const float mo = Mrow[t];
            const float mn = fmaxf(mo, mt);
            Alpha[t] = __expf(mo - mn);
            Mrow[t]  = mn;
        }
        __syncthreads();

        const float mn = Mrow[lane];
        float ps = 0.0f;
#pragma unroll
        for (int c2 = 0; c2 < 4; ++c2) {
            const float p0 = __expf(Sls[lane][wid * 8 + 2 * c2]     - mn);
            const float p1 = __expf(Sls[lane][wid * 8 + 2 * c2 + 1] - mn);
            *reinterpret_cast<v2bf*>(&Pls[lane][wid * 8 + 2 * c2]) = pack2(p0, p1);
            ps += p0 + p1;
        }
        Psum[lane][wid] = ps;
        __syncthreads();

        if (t < 32) {
            float l = Alpha[t] * Lrow[t];
#pragma unroll
            for (int g = 0; g < 8; ++g) l += Psum[t][g];
            Lrow[t] = l;
        }

        // ---- rescale running O by alpha[row] ----
#pragma unroll
        for (int r = 0; r < 8; ++r) {
            const float al = Alpha[wr * 16 + r + hi * 8];
#pragma unroll
            for (int dsu = 0; dsu < 8; ++dsu) o[dsu][r] = o[dsu][r] * al;
        }

        // ---- O += P(16x64) @ K(64x128 slice): A from LDS, B from Kt ----
#pragma unroll
        for (int k2 = 0; k2 < 2; ++k2) {
            v16bf a = load_a_tile(&Pls[wr * 16 + lm][k2 * 32], lane);
#pragma unroll
            for (int dsu = 0; dsu < 8; ++dsu) {
                const bf16_t* krow =
                    Ktb + (size_t)(wc * 128 + dsu * 16 + lm) * SEQ_S + jb + k2 * 32;
                v16bf bm = load_b_tile(krow, lane);
                o[dsu] = wmma_bf16(a, bm, o[dsu]);
            }
        }
    }

    __syncthreads();

    // ---- epilogue: O /= l, store to out[S, B, D] (f32) ----
#pragma unroll
    for (int r = 0; r < 8; ++r) {
        const float linv = 1.0f / Lrow[wr * 16 + r + hi * 8];
        const int srow   = rowbase + wr * 16 + r + hi * 8;
#pragma unroll
        for (int dsu = 0; dsu < 8; ++dsu) {
            const int col = wc * 128 + dsu * 16 + lm;
            out[((size_t)srow * SEQ_B + b) * SEQ_D + col] = o[dsu][r] * linv;
        }
    }
}

// ---------------------------------------------------------------------------
extern "C" void kernel_launch(void* const* d_in, const int* in_sizes, int n_in,
                              void* d_out, int out_size, void* d_ws, size_t ws_size,
                              hipStream_t stream) {
    (void)in_sizes; (void)n_in; (void)out_size; (void)ws_size;
    const float* seq = (const float*)d_in[0];
    const float* Wv  = (const float*)d_in[1];
    const float* bv  = (const float*)d_in[2];
    const float* Wq  = (const float*)d_in[3];
    const float* bq  = (const float*)d_in[4];
    const float* Wk  = (const float*)d_in[5];
    const float* bk  = (const float*)d_in[6];
    float* out = (float*)d_out;

    const size_t mat_bytes = (size_t)SEQ_B * SEQ_S * SEQ_D * sizeof(bf16_t); // 32 MB
    bf16_t* Xbf  = (bf16_t*)d_ws;
    bf16_t* Vbf  = (bf16_t*)((char*)d_ws + 1 * mat_bytes);
    bf16_t* Qbf  = (bf16_t*)((char*)d_ws + 2 * mat_bytes);
    bf16_t* Ktbf = (bf16_t*)((char*)d_ws + 3 * mat_bytes);
    bf16_t* Wbf  = (bf16_t*)((char*)d_ws + 4 * mat_bytes);  // 1.5 MB

    const unsigned n_seq8 = (unsigned)((size_t)SEQ_S * SEQ_B * SEQ_D / 8);   // 2M
    cvt_seq_kernel<<<n_seq8 / 256, 256, 0, stream>>>(seq, Xbf);

    const unsigned n_w8 = 3u * SEQ_D * SEQ_D / 8;                            // 98304
    cvt_w_kernel<<<n_w8 / 256, 256, 0, stream>>>(Wv, Wq, Wk, Wbf);

    dim3 g1(SEQ_D / 64, SEQ_S / 64, SEQ_B * 3), b1(128);
    proj_kernel<<<g1, b1, 0, stream>>>(Xbf, Wbf, bv, bq, bk, Vbf, Qbf, Ktbf);

    dim3 g2(SEQ_S / 32, SEQ_B), b2(256);
    attn_kernel<<<g2, b2, 0, stream>>>(Vbf, Qbf, Ktbf, out);
}
